// RmMultiHeadAttention_5686536699964
// MI455X (gfx1250) — matrix operand order repaired and checked by reading, not compiled, not run on hardware
//
#include <hip/hip_runtime.h>
#include <hip/hip_bf16.h>

// ---------------------------------------------------------------------------
// Fused RmMultiHeadAttention for MI455X (gfx1250, wave32, WMMA f16 16x16x32).
//  Stage 1: transpose+convert weights to f16 Wt[n][k]  (B-frag contiguous)
//  Stage 2: f16-WMMA GEMM projections -> eq/ek [b,h,s,d], evT [b,h,d,s]
//  Stage 3: causal flash attention (full-softmax+mask+renorm == causal softmax)
//  Stage 4: f16-WMMA GEMM x w_m -> fp32 out
//  f16 tile staging into LDS uses the Tensor Data Mover (tensor_load_to_lds +
//  s_wait_tensorcnt) when the builtin is available; sync staging otherwise.
// ---------------------------------------------------------------------------

typedef __attribute__((ext_vector_type(16))) _Float16 v16h;
typedef __attribute__((ext_vector_type(8)))  _Float16 v8h;
typedef __attribute__((ext_vector_type(8)))  float    v8f;
typedef __attribute__((ext_vector_type(4)))  unsigned int v4u;
typedef __attribute__((ext_vector_type(8)))  int      v8i;
typedef __attribute__((ext_vector_type(4)))  int      v4i;

#define B_  4
#define S_  2048
#define H_  1024
#define NH_ 16
#define DH_ 64
#define BS_ (B_ * S_)

#define TM 128
#define TN 128
#define TK 32

#if defined(__has_builtin)
#if __has_builtin(__builtin_amdgcn_tensor_load_to_lds) && \
    __has_builtin(__builtin_amdgcn_s_wait_tensorcnt)
#define HAVE_TDM 1
#endif
#endif
#ifndef HAVE_TDM
#define HAVE_TDM 0
#endif

__device__ inline v8f vzero8() {
  v8f z;
#pragma unroll
  for (int i = 0; i < 8; ++i) z[i] = 0.0f;
  return z;
}

__device__ inline v16h cat8(v8h lo, v8h hi) {
  return __builtin_shufflevector(lo, hi, 0, 1, 2, 3, 4, 5, 6, 7,
                                 8, 9, 10, 11, 12, 13, 14, 15);
}

__device__ inline v8f wmma_f16(v16h a, v16h b, v8f c) {
  // D = A(16x32 f16) x B(32x16 f16) + C(16x16 f32)
  return __builtin_amdgcn_wmma_f32_16x16x32_f16(
      /*neg_a=*/false, a, /*neg_b=*/false, b,
      /*c_mod=*/(short)0, c, /*reuse_a=*/false, /*reuse_b=*/false);
}

#if HAVE_TDM
// TDM: copy a 2D tile (tile_d1 rows x tile_d0 f16 elems, row stride `stride`
// elems) from global memory into LDS, packed row-major. tile_d1==0 => 1D copy.
// D# layout per CDNA5 ISA ch.8: group0 = {flags, lds_addr, global_addr, type},
// group1 = {wg_mask/data_size, tensor dims, tile dims, dim0 stride}.
// This toolchain exposes the 6-arg builtin:
//   (uint32x4 g0, int32x8 g1, int32x4 g2, int32x4 g3, int32x8 g4, i32 cpol)
__device__ inline void tdm_load_tile(void* lds_dst, const void* gsrc,
                                     unsigned tile_d0, unsigned tile_d1,
                                     unsigned tensor_d0, unsigned stride) {
  unsigned lds_off = (unsigned)(unsigned long long)lds_dst;  // low 32 = LDS off
  unsigned long long ga = (unsigned long long)gsrc;
  v4u g0;
  g0[0] = 1u;                                    // count=1, user descriptor
  g0[1] = lds_off;                               // lds_addr (bytes)
  g0[2] = (unsigned)ga;                          // global_addr[31:0]
  g0[3] = (unsigned)((ga >> 32) & 0x01ffffffu)   // global_addr[56:32]
          | (2u << 30);                          // type = 2 ("image")
  v8i g1;
  g1[0] = (int)(1u << 16);                       // wg_mask=0, data_size=2B
  g1[1] = (int)((tensor_d0 & 0xffffu) << 16);    // [47:32]=0, td0[15:0]
  g1[2] = (int)(((tensor_d0 >> 16) & 0xffffu)    // td0[31:16]
                | (0xffffu << 16));              // tensor_dim1 lo = 0xffff
  g1[3] = (int)((tile_d0 & 0xffffu) << 16);      // td1 hi=0, tile_dim0
  g1[4] = (int)(tile_d1 & 0xffffu);              // tile_dim1 (0 => 1D), tile_dim2=0
  g1[5] = (int)stride;                           // tensor_dim0_stride[31:0]
  g1[6] = 0;                                     // stride[47:32]=0, dim1_stride lo
  g1[7] = 0;
  v4i z4 = {0, 0, 0, 0};
  v8i z8 = {0, 0, 0, 0, 0, 0, 0, 0};
  __builtin_amdgcn_tensor_load_to_lds(g0, g1, z4, z4, z8, 0);
}
#endif

// ---------------------------------------------------------------------------
// Stage 1: Wt[n][k] = (f16) W[k][n]     (block 32x8, grid (H/32, H/32))
// ---------------------------------------------------------------------------
__global__ void wt_transpose_f16(const float* __restrict__ w,
                                 _Float16* __restrict__ wt) {
  __shared__ float tile[32][33];
  int tx = threadIdx.x, ty = threadIdx.y;
  int x0 = blockIdx.x * 32;  // n
  int y0 = blockIdx.y * 32;  // k
#pragma unroll
  for (int j = 0; j < 32; j += 8)
    tile[ty + j][tx] = w[(size_t)(y0 + ty + j) * H_ + x0 + tx];
  __syncthreads();
#pragma unroll
  for (int j = 0; j < 32; j += 8)
    wt[(size_t)(x0 + ty + j) * H_ + y0 + tx] = (_Float16)tile[tx][ty + j];
}

// ---------------------------------------------------------------------------
// Stage 2/4: C[M,N] = A[M,1024] * Wt^T, tiled WMMA GEMM.
//  mode 0/1: store f16 into [b,h,s,d]   (eq / ek)
//  mode 2  : store f16 into [b,h,d,s]   (evT)
//  mode 3  : store fp32 row-major into d_out
// ---------------------------------------------------------------------------
__global__ __launch_bounds__(256)
void gemm16(const void* __restrict__ Av, int aIsF32,
            const _Float16* __restrict__ Bt,
            void* __restrict__ outv, int mode) {
  alignas(32) __shared__ _Float16 lA[TM * TK];  // 8 KB
  alignas(32) __shared__ _Float16 lB[TN * TK];  // 8 KB

  const int t = threadIdx.x;
  const int m0 = blockIdx.x * TM;
  const int n0 = blockIdx.y * TN;
  const int wid = t >> 5, l = t & 31;
  const int wm = wid >> 1, wn = wid & 1;       // 4x2 wave grid
  const int arow = l & 15;
  const int akb = (l < 16) ? 0 : 16;           // A frag K base
  const int bc = l & 15;
  const int bkb = (l < 16) ? 0 : 16;           // B frag K base
  const int g = (l >= 16) ? 1 : 0;

  const float*    A32 = (const float*)Av;
  const _Float16* A16 = (const _Float16*)Av;

  v8f acc[2][4];
#pragma unroll
  for (int i = 0; i < 2; ++i)
#pragma unroll
    for (int j = 0; j < 4; ++j) acc[i][j] = vzero8();

  for (int k0 = 0; k0 < H_; k0 += TK) {
    __syncthreads();
    if (aIsF32) {
      int r = t >> 3, c = (t & 7) * 4;
#pragma unroll
      for (int p = 0; p < 4; ++p) {
        float4 v = *(const float4*)&A32[(size_t)(m0 + r + p * 32) * H_ + k0 + c];
        _Float16* d = &lA[(r + p * 32) * TK + c];
        d[0] = (_Float16)v.x; d[1] = (_Float16)v.y;
        d[2] = (_Float16)v.z; d[3] = (_Float16)v.w;
      }
    }
#if HAVE_TDM
    else if (wid == 0) {
      // 2D TDM tile: TM rows x TK halfs, row stride H_
      tdm_load_tile(lA, &A16[(size_t)m0 * H_ + k0], TK, TM, H_, H_);
    }
    if (wid == 0) {
      tdm_load_tile(lB, &Bt[(size_t)n0 * H_ + k0], TK, TN, H_, H_);
      __builtin_amdgcn_s_wait_tensorcnt(0);
    }
#else
    else {
      int r = t >> 2, c = (t & 3) * 8;
#pragma unroll
      for (int p = 0; p < 2; ++p)
        *(v8h*)&lA[(r + p * 64) * TK + c] =
            *(const v8h*)&A16[(size_t)(m0 + r + p * 64) * H_ + k0 + c];
    }
    {
      int r = t >> 2, c = (t & 3) * 8;
#pragma unroll
      for (int p = 0; p < 2; ++p)
        *(v8h*)&lB[(r + p * 64) * TK + c] =
            *(const v8h*)&Bt[(size_t)(n0 + r + p * 64) * H_ + k0 + c];
    }
#endif
    if (k0 + TK < H_)  // hint next B tile toward L2 (global_prefetch_b8)
      __builtin_prefetch(&Bt[(size_t)(n0 + (t >> 1)) * H_ + k0 + TK + (t & 1) * 16],
                         0, 1);
    __syncthreads();

    v16h af[2], bf[4];
#pragma unroll
    for (int i = 0; i < 2; ++i) {
      const _Float16* p = &lA[(wm * 32 + i * 16 + arow) * TK + akb];
      af[i] = cat8(*(const v8h*)p, *(const v8h*)(p + 8));
    }
#pragma unroll
    for (int j = 0; j < 4; ++j)
      bf[j] = *(const v16h*)&lB[(wn * 64 + j * 16 + bc) * TK + bkb];
#pragma unroll
    for (int i = 0; i < 2; ++i)
#pragma unroll
      for (int j = 0; j < 4; ++j)
        acc[i][j] = wmma_f16(af[i], bf[j], acc[i][j]);
  }

  // epilogue: C layout -> VGPR r: row r + 8*(lane>=16), col lane&15
#pragma unroll
  for (int i = 0; i < 2; ++i)
#pragma unroll
    for (int j = 0; j < 4; ++j)
#pragma unroll
      for (int r = 0; r < 8; ++r) {
        int m = m0 + wm * 32 + i * 16 + r + 8 * g;
        int n = n0 + wn * 64 + j * 16 + bc;
        float val = acc[i][j][r];
        if (mode == 3) {
          ((float*)outv)[(size_t)m * H_ + n] = val;
        } else {
          int b = m >> 11, s = m & 2047;
          int h = n >> 6,  d = n & 63;
          size_t addr = (mode == 2)
              ? (((size_t)(b * NH_ + h) * DH_ + d) * S_ + s)
              : (((size_t)(b * NH_ + h) * S_ + s) * DH_ + d);
          ((_Float16*)outv)[addr] = (_Float16)val;
        }
      }
}

// ---------------------------------------------------------------------------
// Stage 3: causal flash attention.
//  grid (S/128, NH, B), 256 threads (8 waves). Wave w owns q rows q0..q0+15.
//  Per 32-key block: TDM-stage K/V^T, 4 score WMMAs + online softmax + 4 P*V.
// ---------------------------------------------------------------------------
__global__ __launch_bounds__(256)
void attn_fwd(const _Float16* __restrict__ eq,
              const _Float16* __restrict__ ek,
              const _Float16* __restrict__ evT,
              _Float16* __restrict__ O) {
  alignas(32) __shared__ _Float16 sk[32 * 64];      // [key][d]    4 KB
  alignas(32) __shared__ _Float16 sv[64 * 32];      // [d][key]    4 KB
  alignas(32) __shared__ _Float16 sp[8 * 16 * 32];  // per-wave P  8 KB

  const int b = blockIdx.z, h = blockIdx.y;
  const int q0wg = blockIdx.x * 128;
  const _Float16* eqh = eq  + (size_t)(b * NH_ + h) * S_ * DH_;
  const _Float16* ekh = ek  + (size_t)(b * NH_ + h) * S_ * DH_;
  const _Float16* evh = evT + (size_t)(b * NH_ + h) * DH_ * S_;

  const int t = threadIdx.x, wid = t >> 5, l = t & 31;
  const int q0 = q0wg + wid * 16;
  const int arow = l & 15, akb = (l < 16) ? 0 : 16;
  const int bc = l & 15,  bkb = (l < 16) ? 0 : 16;
  const int g = (l >= 16) ? 1 : 0;
  _Float16* pw = &sp[wid * 512];

  // Q fragments for this wave's 16 rows (reused across the whole k loop)
  v16h aq[2];
#pragma unroll
  for (int dc = 0; dc < 2; ++dc) {
    const _Float16* p = eqh + (size_t)(q0 + arow) * DH_ + dc * 32;
    aq[dc] = cat8(*(const v8h*)(p + akb), *(const v8h*)(p + akb + 8));
  }

  v8f o[4];
#pragma unroll
  for (int j = 0; j < 4; ++j) o[j] = vzero8();
  float mrow[8], lrow[8];
#pragma unroll
  for (int r = 0; r < 8; ++r) { mrow[r] = -1e30f; lrow[r] = 0.0f; }

  const int nkb = (q0wg >> 5) + 4;  // causal: keys up to q0wg+127
  for (int kb = 0; kb < nkb; ++kb) {
    const int kp0 = kb * 32;
    __syncthreads();
#if HAVE_TDM
    if (wid == 0) {
      // K tile is 32 contiguous rows of 64 halfs -> single 1D TDM copy
      tdm_load_tile(sk, &ekh[(size_t)kp0 * DH_], 32 * 64, 0, 32 * 64, 0);
      // V^T tile: 64 rows x 32 halfs, row stride S_
      tdm_load_tile(sv, &evh[kp0], 32, 64, S_, S_);
      __builtin_amdgcn_s_wait_tensorcnt(0);
    }
#else
    *(v8h*)&sk[(t >> 3) * 64 + (t & 7) * 8] =
        *(const v8h*)&ekh[(size_t)(kp0 + (t >> 3)) * DH_ + (t & 7) * 8];
    *(v8h*)&sv[(t >> 2) * 32 + (t & 3) * 8] =
        *(const v8h*)&evh[(size_t)(t >> 2) * S_ + kp0 + (t & 3) * 8];
#endif
    __syncthreads();
    if (kp0 > q0 + 15) continue;  // block fully masked for this wave

    // scores: S[16q x 32k], contract over dh=64
    v8f s0 = vzero8(), s1 = vzero8();
#pragma unroll
    for (int dc = 0; dc < 2; ++dc) {
      v16h b0 = *(const v16h*)&sk[(0  + bc) * 64 + dc * 32 + bkb];
      v16h b1 = *(const v16h*)&sk[(16 + bc) * 64 + dc * 32 + bkb];
      s0 = wmma_f16(aq[dc], b0, s0);
      s1 = wmma_f16(aq[dc], b1, s1);
    }

    // online softmax (scale 1/sqrt(64)=0.125, causal mask, row reductions)
    float cf[8];
#pragma unroll
    for (int r = 0; r < 8; ++r) {
      int qpos = q0 + r + 8 * g;
      float x0 = s0[r] * 0.125f;
      float x1 = s1[r] * 0.125f;
      if (kp0 + bc      > qpos) x0 = -1e30f;
      if (kp0 + 16 + bc > qpos) x1 = -1e30f;
      float mx = fmaxf(x0, x1);
      mx = fmaxf(mx, __shfl_xor(mx, 1, 32));
      mx = fmaxf(mx, __shfl_xor(mx, 2, 32));
      mx = fmaxf(mx, __shfl_xor(mx, 4, 32));
      mx = fmaxf(mx, __shfl_xor(mx, 8, 32));
      float mnew = fmaxf(mrow[r], mx);
      float p0 = __expf(x0 - mnew);
      float p1 = __expf(x1 - mnew);
      float rs = p0 + p1;
      rs += __shfl_xor(rs, 1, 32);
      rs += __shfl_xor(rs, 2, 32);
      rs += __shfl_xor(rs, 4, 32);
      rs += __shfl_xor(rs, 8, 32);
      float c = __expf(mrow[r] - mnew);
      lrow[r] = lrow[r] * c + rs;
      mrow[r] = mnew;
      cf[r] = c;
      // C layout -> P row-major [16x32] in per-wave LDS scratch
      pw[(r + 8 * g) * 32 + bc]      = (_Float16)p0;
      pw[(r + 8 * g) * 32 + 16 + bc] = (_Float16)p1;
    }

    // rescale O by exp(m_old - m_new), per row
#pragma unroll
    for (int j = 0; j < 4; ++j)
#pragma unroll
      for (int r = 0; r < 8; ++r) o[j][r] *= cf[r];

    // reload P as A-fragment (wave-local LDS round trip = C->A transpose)
    v16h pa;
    {
      const _Float16* p = &pw[arow * 32 + akb];
      pa = cat8(*(const v8h*)p, *(const v8h*)(p + 8));
    }
#pragma unroll
    for (int j = 0; j < 4; ++j) {
      v16h bv = *(const v16h*)&sv[(j * 16 + bc) * 32 + bkb];
      o[j] = wmma_f16(pa, bv, o[j]);
    }
  }

  // normalize and store O as f16 [b,s,h*64+d] for the final GEMM
#pragma unroll
  for (int r = 0; r < 8; ++r) lrow[r] = 1.0f / lrow[r];
#pragma unroll
  for (int j = 0; j < 4; ++j)
#pragma unroll
    for (int r = 0; r < 8; ++r) {
      int qpos = q0 + r + 8 * g;
      int d = j * 16 + bc;
      O[(size_t)(b * S_ + qpos) * H_ + h * DH_ + d] =
          (_Float16)(o[j][r] * lrow[r]);
    }
}

// ---------------------------------------------------------------------------
extern "C" void kernel_launch(void* const* d_in, const int* in_sizes, int n_in,
                              void* d_out, int out_size, void* d_ws,
                              size_t ws_size, hipStream_t stream) {
  (void)in_sizes; (void)n_in; (void)out_size; (void)ws_size;
  const float* q   = (const float*)d_in[0];
  const float* k   = (const float*)d_in[1];
  const float* v   = (const float*)d_in[2];
  const float* w_q = (const float*)d_in[3];
  const float* w_k = (const float*)d_in[4];
  const float* w_v = (const float*)d_in[5];
  const float* w_m = (const float*)d_in[6];

  char* ws = (char*)d_ws;
  size_t off = 0;
  const size_t wbytes = (size_t)H_ * H_ * sizeof(_Float16);             // 2 MB
  const size_t abytes = (size_t)B_ * NH_ * S_ * DH_ * sizeof(_Float16); // 16 MB
  _Float16* wq_t = (_Float16*)(ws + off); off += wbytes;
  _Float16* wk_t = (_Float16*)(ws + off); off += wbytes;
  _Float16* wv_t = (_Float16*)(ws + off); off += wbytes;
  _Float16* wm_t = (_Float16*)(ws + off); off += wbytes;
  _Float16* eqb  = (_Float16*)(ws + off); off += abytes;  // [b,h,s,d]
  _Float16* ekb  = (_Float16*)(ws + off); off += abytes;  // [b,h,s,d]
  _Float16* evtb = (_Float16*)(ws + off); off += abytes;  // [b,h,d,s]
  _Float16* obuf = (_Float16*)(ws + off); off += abytes;  // [b,s,h*64+d]

  dim3 tb(32, 8), tg(H_ / 32, H_ / 32);
  wt_transpose_f16<<<tg, tb, 0, stream>>>(w_q, wq_t);
  wt_transpose_f16<<<tg, tb, 0, stream>>>(w_k, wk_t);
  wt_transpose_f16<<<tg, tb, 0, stream>>>(w_v, wv_t);
  wt_transpose_f16<<<tg, tb, 0, stream>>>(w_m, wm_t);

  dim3 gg(BS_ / TM, H_ / TN);
  gemm16<<<gg, 256, 0, stream>>>(q, 1, wq_t, eqb, 0);
  gemm16<<<gg, 256, 0, stream>>>(k, 1, wk_t, ekb, 1);
  gemm16<<<gg, 256, 0, stream>>>(v, 1, wv_t, evtb, 2);

  attn_fwd<<<dim3(S_ / 128, NH_, B_), 256, 0, stream>>>(eqb, ekb, evtb, obuf);

  gemm16<<<gg, 256, 0, stream>>>(obuf, 0, wm_t, d_out, 3);
}
